// ExpertChoiceGate_83708912599441
// MI455X (gfx1250) — matrix operand order repaired and checked by reading, not compiled, not run on hardware
//
#include <hip/hip_runtime.h>
#include <hip/hip_bf16.h>

typedef __attribute__((ext_vector_type(16))) __bf16 v16bf;
typedef __attribute__((ext_vector_type(8)))  float  v8f;
typedef __attribute__((ext_vector_type(4)))  float  v4f;

#define BB   4
#define SS   4096
#define DD   4096
#define EE   64
#define MM   (BB * SS)   // 16384 tokens
#define TOPK (SS / EE)   // 64
#define PFD  512         // A-stream prefetch distance in floats (2KB, ~16 K-steps)

// ---------------------------------------------------------------------------
// Kernel 0: convert w_gate [E, D] fp32 -> bf16 (row-major, same layout).
// ---------------------------------------------------------------------------
__global__ void cvt_wgate_kernel(const float* __restrict__ w,
                                 __hip_bfloat16* __restrict__ wb) {
    int i = blockIdx.x * blockDim.x + threadIdx.x;
    if (i < EE * DD) wb[i] = __float2bfloat16(w[i]);
}

// ---------------------------------------------------------------------------
// Kernel 1: x_gated[M, E] = x[M, D] * w_gate^T via v_wmma_f32_16x16x32_bf16.
// Block = 128 threads (4 waves); each wave computes a 16x64 tile (4 x v8f acc).
// A fragment (16x32 bf16): lane L<16 -> row M=L, K = 0..15 (contiguous);
//                          lane L>=16 -> row M=L-16, K = 16..31 (contiguous).
// B fragment (32x16 bf16): lane L<16 -> col N=L, K = 0..15 (contiguous);
//                          lane L>=16 -> col N=L-16, K = 16..31 (contiguous).
// HBM-bound on the fp32 A stream (256 MB, ~11us at 23.3 TB/s). Unconditional
// speculative prefetch (locality 3 -> near scope) keeps the stream ahead;
// overshoot past the row end is silently dropped / warms the next row.
// ---------------------------------------------------------------------------
__global__ void __launch_bounds__(128)
gate_gemm_kernel(const float* __restrict__ x,
                 const __hip_bfloat16* __restrict__ wb,
                 float* __restrict__ xg) {
    const int lane    = threadIdx.x & 31;
    const int wave    = threadIdx.x >> 5;
    const int laneLo  = lane & 15;
    const int hi      = lane >> 4;                 // 0 or 1
    const int rowBase = blockIdx.x * 64 + wave * 16;

    const float* arow = x + (size_t)(rowBase + laneLo) * DD;
    const int ksel = hi * 16;

    v8f acc0 = {}, acc1 = {}, acc2 = {}, acc3 = {};

    for (int k0 = 0; k0 < DD; k0 += 32) {
        // ---- speculative prefetch of the A stream (no guard: keeps loop clean)
        __builtin_prefetch(arow + k0 + ksel + PFD, 0, 3);

        // ---- A fragment: 4 x 16B loads, fp32 -> bf16 pack in registers
        v4f a0 = *(const v4f*)(arow + k0 + ksel);
        v4f a1 = *(const v4f*)(arow + k0 + ksel + 4);
        v4f a2 = *(const v4f*)(arow + k0 + ksel + 8);
        v4f a3 = *(const v4f*)(arow + k0 + ksel + 12);
        v16bf afrag;
#pragma unroll
        for (int i = 0; i < 4; ++i) {
            afrag[i]      = (__bf16)a0[i];
            afrag[4 + i]  = (__bf16)a1[i];
            afrag[8 + i]  = (__bf16)a2[i];
            afrag[12 + i] = (__bf16)a3[i];
        }

        // ---- B fragments: 16 consecutive K values per lane => 32B contiguous
        const __hip_bfloat16* bbase =
            wb + (size_t)laneLo * DD + (size_t)(k0 + hi * 16);
        v16bf b0 = *(const v16bf*)(bbase + (size_t)0  * DD);
        v16bf b1 = *(const v16bf*)(bbase + (size_t)16 * DD);
        v16bf b2 = *(const v16bf*)(bbase + (size_t)32 * DD);
        v16bf b3 = *(const v16bf*)(bbase + (size_t)48 * DD);

        acc0 = __builtin_amdgcn_wmma_f32_16x16x32_bf16(false, afrag, false, b0,
                                                       (short)0, acc0, false, false);
        acc1 = __builtin_amdgcn_wmma_f32_16x16x32_bf16(false, afrag, false, b1,
                                                       (short)0, acc1, false, false);
        acc2 = __builtin_amdgcn_wmma_f32_16x16x32_bf16(false, afrag, false, b2,
                                                       (short)0, acc2, false, false);
        acc3 = __builtin_amdgcn_wmma_f32_16x16x32_bf16(false, afrag, false, b3,
                                                       (short)0, acc3, false, false);
    }

    // ---- store D: VGPR j -> row M = j + 8*hi, col N = laneLo (+16t per frag)
#pragma unroll
    for (int j = 0; j < 8; ++j) {
        const int m = rowBase + j + hi * 8;
        float* o = xg + (size_t)m * EE + laneLo;
        o[0]  = acc0[j];
        o[16] = acc1[j];
        o[32] = acc2[j];
        o[48] = acc3[j];
    }
}

// ---------------------------------------------------------------------------
// Kernel 2: per-token softmax stats (row max and sum of exp over E=64).
// ---------------------------------------------------------------------------
__global__ void rowstats_kernel(const float* __restrict__ xg,
                                float* __restrict__ rmax,
                                float* __restrict__ rsum) {
    int t = blockIdx.x * blockDim.x + threadIdx.x;
    if (t >= MM) return;
    const v4f* p = (const v4f*)(xg + (size_t)t * EE);
    v4f v[16];
    float m = -INFINITY;
#pragma unroll
    for (int i = 0; i < 16; ++i) {
        v[i] = p[i];
        m = fmaxf(m, fmaxf(fmaxf(v[i][0], v[i][1]), fmaxf(v[i][2], v[i][3])));
    }
    float s = 0.f;
#pragma unroll
    for (int i = 0; i < 16; ++i) {
        s += __expf(v[i][0] - m) + __expf(v[i][1] - m) +
             __expf(v[i][2] - m) + __expf(v[i][3] - m);
    }
    rmax[t] = m;
    rsum[t] = s;
}

// ---------------------------------------------------------------------------
// Kernel 3: expert-choice top-64 over the sequence dim for each (b, e) pair,
// then scatter softmax probs into the (pre-zeroed) gate output.
// Iterative block-argmax with lowest-index tie-break (matches lax.top_k).
// ---------------------------------------------------------------------------
__global__ void __launch_bounds__(256)
topk_kernel(const float* __restrict__ xg,
            const float* __restrict__ rmax,
            const float* __restrict__ rsum,
            float* __restrict__ gate) {
    __shared__ float vals[SS];
    __shared__ float cval[256];
    __shared__ int   cidx[256];

    const int b   = blockIdx.x >> 6;   // / EE
    const int e   = blockIdx.x & 63;   // % EE
    const int tid = threadIdx.x;

    // gather this expert's logit column across the sequence
#pragma unroll
    for (int i = 0; i < 16; ++i) {
        int s = tid + i * 256;
        vals[s] = xg[((size_t)(b * SS + s)) * EE + e];
    }
    __syncthreads();

    for (int iter = 0; iter < TOPK; ++iter) {
        // local argmax over a contiguous 16-element chunk (lowest index wins)
        float bv = -INFINITY;
        int   bi = 0;
        const int base = tid * 16;
#pragma unroll
        for (int i = 0; i < 16; ++i) {
            float v = vals[base + i];
            if (v > bv) { bv = v; bi = base + i; }
        }
        cval[tid] = bv;
        cidx[tid] = bi;
        __syncthreads();

        // tree reduction, tie-break toward lower sequence index
        for (int off = 128; off > 0; off >>= 1) {
            if (tid < off) {
                float ov = cval[tid + off];
                int   oi = cidx[tid + off];
                if (ov > cval[tid] || (ov == cval[tid] && oi < cidx[tid])) {
                    cval[tid] = ov;
                    cidx[tid] = oi;
                }
            }
            __syncthreads();
        }

        if (tid == 0) {
            const int s   = cidx[0];
            const int tok = b * SS + s;
            const float prob = __expf(cval[0] - rmax[tok]) / rsum[tok];
            gate[(size_t)tok * EE + e] = prob;
            vals[s] = -INFINITY;   // remove from future rounds
        }
        __syncthreads();
    }
}

// ---------------------------------------------------------------------------
extern "C" void kernel_launch(void* const* d_in, const int* in_sizes, int n_in,
                              void* d_out, int out_size, void* d_ws, size_t ws_size,
                              hipStream_t stream) {
    const float* x = (const float*)d_in[0];   // [B, S, D]
    const float* w = (const float*)d_in[1];   // [E, D]

    float* gate = (float*)d_out;                        // output 0: gate*mask
    float* xg   = (float*)d_out + (size_t)MM * EE;      // output 1: x_gated

    // workspace: bf16 weights (512KB) | rowmax (64KB) | rowsum (64KB)
    __hip_bfloat16* wb = (__hip_bfloat16*)d_ws;
    float* rmax = (float*)((char*)d_ws + (size_t)EE * DD * sizeof(__hip_bfloat16));
    float* rsum = rmax + MM;

    cvt_wgate_kernel<<<(EE * DD + 255) / 256, 256, 0, stream>>>(w, wb);
    gate_gemm_kernel<<<MM / 64, 128, 0, stream>>>(x, wb, xg);
    rowstats_kernel<<<MM / 256, 256, 0, stream>>>(xg, rmax, rsum);
    hipMemsetAsync(gate, 0, (size_t)MM * EE * sizeof(float), stream);
    topk_kernel<<<BB * EE, 256, 0, stream>>>(xg, rmax, rsum, gate);
}
